// ConvDU_61942018343387
// MI455X (gfx1250) — matrix operand order repaired and checked by reading, not compiled, not run on hardware
//
#include <hip/hip_runtime.h>

// ---------------------------------------------------------------------------
// Recurrent row-scan of conv1d(C=256, K=9) + ReLU + residual, on MI455X.
// Each scan step = GEMM  M=256(co) x N=1024(n,w) x K=2304(ci,k) using
// v_wmma_f32_16x16x32_bf16. The bf16 carry row is staged into LDS by the
// Tensor Data Mover (tensor_load_to_lds, TENSORcnt), with hardware row
// padding producing the bank-rotated 544B LDS row stride.
// ---------------------------------------------------------------------------

typedef __attribute__((ext_vector_type(16))) __bf16   v16bf;
typedef __attribute__((ext_vector_type(8)))  __bf16   v8bf;
typedef __attribute__((ext_vector_type(8)))  float    v8f;
typedef __attribute__((ext_vector_type(4)))  unsigned u32x4;
typedef __attribute__((ext_vector_type(4)))  int      i32x4;
typedef __attribute__((ext_vector_type(8)))  int      i32x8;

#define CH        256      // channels
#define KW        9        // conv taps
#define NB        8        // batch
#define H         128
#define W         128
#define KRED      (CH*KW)            // 2304, reduction r = k*256 + ci
#define NCHUNK    (KRED/32)          // 72 K-chunks of 32
#define LDS_ROWS  72                 // 64 w + 8 halo
#define LDS_STRIDE 272               // 256 ci + 16 pad bf16: 544B row (8-bank rotate)
#define A_CHUNK_ELEMS (16*32*16)     // one K-chunk of A: 16 m-tiles x 32 lanes x 16
#define ABUF_BYTES ((size_t)NCHUNK * A_CHUNK_ELEMS * 2)   // 0x120000
#define CARRY_ELEMS ((size_t)NB * W * CH)                 // 262144 bf16 = 512KB

__device__ __forceinline__ __bf16 to_bf16(float f) { return (__bf16)f; }

// ---------------------------------------------------------------------------
// One-time: swizzle weight (co,ci,k) f32 -> bf16 A-fragments in the exact
// per-lane K order shared with step_kernel's B loads (contiguous K, not the ISA table order):
//   lane<16 : K = {0..15}, lane>=16 : K = {16..31}, M = lane%16
// Stored as Abuf[((q*16 + mtile)*32 + lane)*16 + j].
// ---------------------------------------------------------------------------
__global__ void build_A_kernel(const float* __restrict__ w, __bf16* __restrict__ Abuf) {
    int idx = blockIdx.x * blockDim.x + threadIdx.x;
    if (idx >= NCHUNK * A_CHUNK_ELEMS) return;
    int j    = idx & 15;
    int lane = (idx >> 4) & 31;
    int mt   = (idx >> 9) & 15;
    int q    = idx >> 13;
    int m      = mt * 16 + (lane & 15);
    int klocal = j + ((lane >= 16) ? 16 : 0);
    int r  = q * 32 + klocal;
    int k  = r >> 8;
    int ci = r & 255;
    Abuf[idx] = to_bf16(w[(m * CH + ci) * KW + k]);
}

// out[:, :, 0, :] = fea[:, :, 0, :]   and   carry0[(n,w),c] = bf16(row 0)
__global__ void copy_row0_kernel(const float* __restrict__ fea, float* __restrict__ out,
                                 __bf16* __restrict__ carry0) {
    int idx = blockIdx.x * blockDim.x + threadIdx.x;     // NB*CH*W
    if (idx >= NB * CH * W) return;
    int w    = idx % W;
    int flat = idx / W;            // n*CH + c
    int c    = flat % CH;
    int n    = flat / CH;
    float v = fea[(size_t)flat * (H * W) + w];
    out[(size_t)flat * (H * W) + w] = v;
    carry0[((size_t)n * W + w) * CH + c] = to_bf16(v);
}

// ---------------------------------------------------------------------------
// One scan step:
//   res = relu(conv1d(carry_src) + bias) + add_src[.,.,h,.]
//   out[.,.,h,.] = res ;  carry_dst[(n,w),c] = bf16(res)
// Grid x = 4 (cg = x>>1 : 128-co group, wg = x&1 : 64-w group), y = batch n.
// Block 256 = 8 wave32: wave = (mgrp 0..3) x (ngrp 0..1) -> 32co x 32w.
// ---------------------------------------------------------------------------
__global__ __launch_bounds__(256)
void step_kernel(float* __restrict__ out, const float* __restrict__ add_src,
                 const __bf16* __restrict__ Abuf, const float* __restrict__ bias,
                 const __bf16* __restrict__ carry_src, __bf16* __restrict__ carry_dst,
                 int h) {
    __shared__ __bf16 lds[LDS_ROWS * LDS_STRIDE];

    const int tid = threadIdx.x;
    const int n   = blockIdx.y;
    const int cg  = blockIdx.x >> 1;          // co base = cg*128
    const int wb  = (blockIdx.x & 1) * 64;    // w base of this tile

    // ---- zero the 4 halo rows (w<0 at left tile, w>=128 at right tile) -----
    {
        const int zrow0 = (wb == 0) ? 0 : 68;
        const int r = tid >> 6;               // 0..3
        const int c = (tid & 63) * 4;         // 0..252, 4 bf16 each
        *(unsigned long long*)&lds[(zrow0 + r) * LDS_STRIDE + c] = 0ull;
    }

    // ---- TDM: one tensor_load_to_lds stages 68 rows x 256 bf16 with pad ----
    // LDS gets rows [lds_row0 .. lds_row0+68) at stride 544B (128dw + 8dw pad).
    if (tid < 32) {
        const int w_start  = (wb == 0) ? 0 : 60;     // global w of first row
        const int lds_row0 = (wb == 0) ? 4 : 0;
        const unsigned lds_addr =
            (unsigned)(uintptr_t)(&lds[0]) + (unsigned)(lds_row0 * LDS_STRIDE * 2);
        const unsigned long long ga =
            (unsigned long long)(uintptr_t)carry_src +
            ((unsigned long long)(n * W + w_start) * CH) * 2ull;

        u32x4 g0;
        g0[0] = 1u;                                   // count=1, no gather
        g0[1] = lds_addr;                             // lds_addr [63:32]
        g0[2] = (unsigned)ga;                         // global_addr[31:0]
        g0[3] = ((unsigned)(ga >> 32) & 0x01FFFFFFu)  // global_addr[56:32]
                | 0x80000000u;                        // type=2 ("image")
        i32x8 g1;
        g1[0] = (int)((1u << 16)        // data_size = 2B
                    | (1u << 20)        // pad_enable
                    | (6u << 22)        // pad_interval: 128 dwords
                    | (7u << 25));      // pad_amount: 8 dwords (32B)
        g1[1] = (int)(256u << 16);      // tensor_dim0[15:0]=256 (abar_addr=0)
        g1[2] = (int)(68u  << 16);      // tensor_dim0[31:16]=0, tensor_dim1[15:0]=68
        g1[3] = (int)(256u << 16);      // tensor_dim1[31:16]=0, tile_dim0=256
        g1[4] = 68;                     // tile_dim1=68, tile_dim2=0
        g1[5] = 256;                    // tensor_dim0_stride[31:0]=256
        g1[6] = 0;                      // stride0[47:32]=0, stride1 lo=0
        g1[7] = 0;
        i32x4 gz4 = {0, 0, 0, 0};
        i32x8 gz8 = {0, 0, 0, 0, 0, 0, 0, 0};
        __builtin_amdgcn_tensor_load_to_lds(g0, g1, gz4, gz4, gz8, 0);
        __builtin_amdgcn_s_wait_tensorcnt(0);
    }
    __syncthreads();

    const int lane = tid & 31;
    const int wave = tid >> 5;
    const int mgrp = wave & 3;                // 4 groups of 32 co
    const int ngrp = wave >> 2;               // 2 groups of 32 w
    const int ln16 = lane & 15;
    const int hi   = lane >> 4;
    const int mt0  = cg * 8 + mgrp * 2;       // first of 2 m-tiles

    const int brow0 = ngrp * 32 + ln16;       // lds w-row (sans +k), n-tile 0
    const int brow1 = brow0 + 16;
    const int cioff = hi * 16;                // upper half-wave: K=16..31 -> ci+16

    v8f acc00 = {}, acc01 = {}, acc10 = {}, acc11 = {};
    const __bf16* Awave = Abuf + ((size_t)mt0 * 32 + lane) * 16;

    #pragma unroll 4
    for (int q = 0; q < NCHUNK; ++q) {
        const int k  = q >> 3;
        const int cb = ((q & 7) << 5) + cioff;

        v16bf a0 = *(const v16bf*)(Awave + (size_t)q * A_CHUNK_ELEMS);
        v16bf a1 = *(const v16bf*)(Awave + (size_t)q * A_CHUNK_ELEMS + 32 * 16);
        v16bf b0 = *(const v16bf*)(&lds[(brow0 + k) * LDS_STRIDE + cb]);
        v16bf b1 = *(const v16bf*)(&lds[(brow1 + k) * LDS_STRIDE + cb]);

        acc00 = __builtin_amdgcn_wmma_f32_16x16x32_bf16(false, a0, false, b0, (short)0, acc00, false, false);
        acc01 = __builtin_amdgcn_wmma_f32_16x16x32_bf16(false, a0, false, b1, (short)0, acc01, false, false);
        acc10 = __builtin_amdgcn_wmma_f32_16x16x32_bf16(false, a1, false, b0, (short)0, acc10, false, false);
        acc11 = __builtin_amdgcn_wmma_f32_16x16x32_bf16(false, a1, false, b1, (short)0, acc11, false, false);
    }

    // ---- epilogue: bias+relu+residual; f32 out row + packed bf16 carry row
    // C/D layout: VGPR v -> M = v + 8*hi, N = ln16.
    v8f accs[2][2] = { { acc00, acc01 }, { acc10, acc11 } };
    #pragma unroll
    for (int mi = 0; mi < 2; ++mi) {
        const int co_base = (mt0 + mi) * 16 + hi * 8;
        #pragma unroll
        for (int ni = 0; ni < 2; ++ni) {
            const int w = wb + ngrp * 32 + ni * 16 + ln16;
            v8bf cv;
            #pragma unroll
            for (int v = 0; v < 8; ++v) {
                const int co = co_base + v;
                const size_t o = (((size_t)n * CH + co) * H + h) * W + w;
                float val = accs[mi][ni][v] + bias[co];
                val = val > 0.0f ? val : 0.0f;
                const float res = val + add_src[o];
                out[o] = res;
                cv[v]  = to_bf16(res);
            }
            *(v8bf*)&carry_dst[((size_t)n * W + w) * CH + co_base] = cv;
        }
    }
}

extern "C" void kernel_launch(void* const* d_in, const int* in_sizes, int n_in,
                              void* d_out, int out_size, void* d_ws, size_t ws_size,
                              hipStream_t stream) {
    const float* fea    = (const float*)d_in[0];
    const float* weight = (const float*)d_in[1];
    const float* bias   = (const float*)d_in[2];
    float*  out   = (float*)d_out;
    __bf16* Abuf  = (__bf16*)d_ws;                                    // 1.125 MB
    __bf16* carry0 = (__bf16*)((char*)d_ws + ABUF_BYTES);             // 512 KB
    __bf16* carry1 = (__bf16*)((char*)d_ws + ABUF_BYTES + CARRY_ELEMS * 2);

    {
        int total = NCHUNK * A_CHUNK_ELEMS;
        build_A_kernel<<<(total + 255) / 256, 256, 0, stream>>>(weight, Abuf);
    }
    {
        int total = NB * CH * W;
        copy_row0_kernel<<<(total + 255) / 256, 256, 0, stream>>>(fea, out, carry0);
    }

    __bf16* cbuf[2] = { carry0, carry1 };
    dim3 grid(4, NB);
    // forward: rows 1..127, carry = buf[(h-1)&1] -> buf[h&1], residual = fea[h]
    for (int hh = 1; hh <= H - 1; ++hh)
        step_kernel<<<grid, 256, 0, stream>>>(out, fea, Abuf, bias,
                                              cbuf[(hh - 1) & 1], cbuf[hh & 1], hh);
    // backward: rows 126..1, carry = buf[(h+1)&1] -> buf[h&1], residual = fwd out[h]
    for (int hh = H - 2; hh >= 1; --hh)
        step_kernel<<<grid, 256, 0, stream>>>(out, out, Abuf, bias,
                                              cbuf[(hh + 1) & 1], cbuf[hh & 1], hh);
}
